// GConvLSTM_61924838473854
// MI455X (gfx1250) — matrix operand order repaired and checked by reading, not compiled, not run on hardware
//
#include <hip/hip_runtime.h>
#include <hip/hip_bf16.h>
#include <math.h>

// ---------------------------------------------------------------------------
// GConvLSTM (ChebConv K=3, 4 gates) for MI455X / gfx1250.
// Pipeline: deg -> norm -> 4x sparse prop (atomics) -> pack A/B to bf16
//           -> WMMA GEMM (N x 384) @ (384 x 256), 32x64 macro-tile per wave
//           -> gate elementwise.
// ---------------------------------------------------------------------------

typedef __attribute__((ext_vector_type(16))) __bf16 v16bf;
typedef __attribute__((ext_vector_type(8)))  float  v8f;

// float -> bf16 (round to nearest even), stored as raw u16
__device__ __forceinline__ unsigned short f2bf(float f) {
    unsigned int u = __float_as_uint(f);
    u += 0x7FFFu + ((u >> 16) & 1u);
    return (unsigned short)(u >> 16);
}

__device__ __forceinline__ float sigmoidf(float x) {
    return 1.0f / (1.0f + expf(-x));
}

// ---- Stage 1: deg[src] += w ------------------------------------------------
__global__ void deg_kernel(const float* __restrict__ w, const int* __restrict__ src,
                           float* __restrict__ deg, int E) {
    int e = blockIdx.x * blockDim.x + threadIdx.x;
    if (e < E) atomicAdd(&deg[src[e]], w[e]);
}

// ---- Stage 2: norm = -dinv[src] * w * dinv[dst] ----------------------------
__global__ void norm_kernel(const int* __restrict__ src, const int* __restrict__ dst,
                            const float* __restrict__ w, const float* __restrict__ deg,
                            float* __restrict__ nrm, int E) {
    int e = blockIdx.x * blockDim.x + threadIdx.x;
    if (e >= E) return;
    float ds = deg[src[e]], dd = deg[dst[e]];
    float a = ds > 0.0f ? rsqrtf(ds) : 0.0f;
    float b = dd > 0.0f ? rsqrtf(dd) : 0.0f;
    nrm[e] = -a * w[e] * b;
}

// ---- Stage 3: out[dst] += nrm * in[src]  (4 threads per edge, 16 ch each) --
__global__ void prop_kernel(const float* __restrict__ in, float* __restrict__ out,
                            const int* __restrict__ src, const int* __restrict__ dst,
                            const float* __restrict__ nrm, int E) {
    int t = blockIdx.x * blockDim.x + threadIdx.x;
    int e  = t >> 2;
    int cg = (t & 3) << 4;          // channel group base (0,16,32,48)
    if (e >= E) return;
    int s = src[e], d = dst[e];
    float nv = nrm[e];
    const float4* ip = (const float4*)(in + (size_t)s * 64 + cg);
    float* op = out + (size_t)d * 64 + cg;
#pragma unroll
    for (int q = 0; q < 4; ++q) {
        float4 v = ip[q];
        atomicAdd(op + q * 4 + 0, nv * v.x);
        atomicAdd(op + q * 4 + 1, nv * v.y);
        atomicAdd(op + q * 4 + 2, nv * v.z);
        atomicAdd(op + q * 4 + 3, nv * v.w);
    }
}

// ---- Stage 4: pack A = [X, T1X, 2*PX-X, H, T1H, 2*PH-H] as bf16, Mpad x 384
__global__ void packA_kernel(const float* __restrict__ X,  const float* __restrict__ T1X,
                             const float* __restrict__ PX, const float* __restrict__ H,
                             const float* __restrict__ T1H, const float* __restrict__ PH,
                             unsigned short* __restrict__ Abf, int N, int Mpad) {
    int idx = blockIdx.x * blockDim.x + threadIdx.x;
    if (idx >= Mpad * 384) return;
    int n = idx / 384;
    int k = idx - n * 384;
    int blk = k >> 6;
    int c   = k & 63;
    float v = 0.0f;
    if (n < N) {
        size_t p = (size_t)n * 64 + c;
        switch (blk) {
            case 0: v = X[p];                       break;
            case 1: v = T1X[p];                     break;
            case 2: v = 2.0f * PX[p] - X[p];        break;
            case 3: v = H[p];                       break;
            case 4: v = T1H[p];                     break;
            default: v = 2.0f * PH[p] - H[p];       break;
        }
    }
    Abf[idx] = f2bf(v);
}

// ---- Stage 5: pack B into WMMA fragment order ------------------------------
// Bpack[((s*16 + t)*32 + lane)*16 + j]:
//   K    = s*32 + (lane>=16 ? 16 : 0) + j
//   Ncol = t*16 + (lane & 15)
// K-block 0..2 -> Wx[g][kb], 3..5 -> Wh[g][kb-3]; g = Ncol/64.
__global__ void packB_kernel(const float* __restrict__ Wx, const float* __restrict__ Wh,
                             unsigned short* __restrict__ Bp) {
    int idx = blockIdx.x * blockDim.x + threadIdx.x;
    if (idx >= 12 * 16 * 32 * 16) return;
    int j    = idx & 15;
    int lane = (idx >> 4) & 31;
    int t    = (idx >> 9) & 15;
    int s    = idx >> 13;
    int K    = s * 32 + ((lane >> 4) << 4) + j;
    int Ncol = t * 16 + (lane & 15);
    int blk = K >> 6, cin = K & 63;
    int g   = Ncol >> 6, co = Ncol & 63;
    float v;
    if (blk < 3) v = Wx[(((size_t)(g * 3 + blk))     * 64 + cin) * 64 + co];
    else         v = Wh[(((size_t)(g * 3 + blk - 3)) * 64 + cin) * 64 + co];
    Bp[idx] = f2bf(v);
}

// ---- Stage 6: WMMA GEMM: Gates[Mpad x 256] = A[Mpad x 384] @ B[384 x 256] --
// One wave per 32(M) x 64(N) macro-tile: 2 A frags + 4 B frags feed 8 WMMAs
// per k-step (12 k-steps of v_wmma_f32_16x16x32_bf16 => 96 wmma / wave).
__global__ void __launch_bounds__(256) gemm_wmma_kernel(
        const unsigned short* __restrict__ Abf,
        const unsigned short* __restrict__ Bp,
        float* __restrict__ G, int totalWaves) {
    const int lane = threadIdx.x & 31;
    const int wave = threadIdx.x >> 5;
    const int tile = blockIdx.x * 8 + wave;
    if (tile >= totalWaves) return;           // wave-uniform guard: EXEC stays all-1
    const int ng     = tile & 3;              // which 64-wide column group (0..3)
    const int tile_m = tile >> 2;             // which 32-row group

    // A fragment rows: lanes 0-15 -> row = base+lane (K offs 0..15)
    //                  lanes 16-31 -> row = base+lane-16 (K offs 16..31)
    const int rbase  = tile_m * 32 + (lane & 15);
    const int ahalf  = (lane >> 4) << 4;      // 0 or 16
    const unsigned short* arow0 = Abf + (size_t)rbase * 384 + ahalf;
    const unsigned short* arow1 = arow0 + (size_t)16 * 384;
    // B fragments for 4 consecutive 16-col tiles of this column group
    const unsigned short* bbase = Bp + ((size_t)(ng * 4) * 32 + lane) * 16;

    v8f acc[2][4];
#pragma unroll
    for (int mi = 0; mi < 2; ++mi)
#pragma unroll
        for (int ni = 0; ni < 4; ++ni)
            acc[mi][ni] = (v8f){0.f, 0.f, 0.f, 0.f, 0.f, 0.f, 0.f, 0.f};

    union Frag { v16bf v; uint4 u[2]; };

#pragma unroll
    for (int s = 0; s < 12; ++s) {
        Frag a0, a1, b[4];
        const unsigned short* ap0 = arow0 + s * 32;
        const unsigned short* ap1 = arow1 + s * 32;
        a0.u[0] = *(const uint4*)(ap0);
        a0.u[1] = *(const uint4*)(ap0 + 8);
        a1.u[0] = *(const uint4*)(ap1);
        a1.u[1] = *(const uint4*)(ap1 + 8);
        const unsigned short* bs = bbase + (size_t)s * (16 * 32 * 16);
#pragma unroll
        for (int ni = 0; ni < 4; ++ni) {
            const unsigned short* bp = bs + (size_t)ni * (32 * 16);
            b[ni].u[0] = *(const uint4*)(bp);
            b[ni].u[1] = *(const uint4*)(bp + 8);
        }
#pragma unroll
        for (int ni = 0; ni < 4; ++ni) {
            acc[0][ni] = __builtin_amdgcn_wmma_f32_16x16x32_bf16(
                false, a0.v, false, b[ni].v, (short)0, acc[0][ni], false, false);
            acc[1][ni] = __builtin_amdgcn_wmma_f32_16x16x32_bf16(
                false, a1.v, false, b[ni].v, (short)0, acc[1][ni], false, false);
        }
    }

    // D layout: VGPR v -> M = v + (lane>=16 ? 8 : 0), N = lane & 15
    const int mb = (lane >> 4) << 3;
    const int cb = lane & 15;
#pragma unroll
    for (int mi = 0; mi < 2; ++mi)
#pragma unroll
        for (int ni = 0; ni < 4; ++ni) {
            float* gout = G + (size_t)(tile_m * 32 + mi * 16 + mb) * 256
                            + ng * 64 + ni * 16 + cb;
#pragma unroll
            for (int v = 0; v < 8; ++v)
                gout[(size_t)v * 256] = acc[mi][ni][v];
        }
}

// ---- Stage 7: gate nonlinearity + state update -----------------------------
__global__ void gates_kernel(const float* __restrict__ G, const float* __restrict__ C,
                             const float* __restrict__ bx, const float* __restrict__ bh,
                             const float* __restrict__ bg, const float* __restrict__ wc,
                             float* __restrict__ Hout, float* __restrict__ Cout, int total) {
    int idx = blockIdx.x * blockDim.x + threadIdx.x;
    if (idx >= total) return;
    int node = idx >> 6;
    int c    = idx & 63;
    const float* g = G + (size_t)node * 256;
    float bi = bx[c]       + bh[c]       + bg[c];
    float bf = bx[64 + c]  + bh[64 + c]  + bg[64 + c];
    float bc = bx[128 + c] + bh[128 + c] + bg[128 + c];
    float bo = bx[192 + c] + bh[192 + c] + bg[192 + c];
    float cv = C[idx];
    float I = sigmoidf(g[c]       + bi + wc[c]       * cv);
    float F = sigmoidf(g[64 + c]  + bf + wc[64 + c]  * cv);
    float T = tanhf  (g[128 + c] + bc);
    float Cn = F * cv + I * T;
    float O = sigmoidf(g[192 + c] + bo + wc[128 + c] * Cn);
    Hout[idx] = O * tanhf(Cn);
    Cout[idx] = Cn;
}

// ---------------------------------------------------------------------------
extern "C" void kernel_launch(void* const* d_in, const int* in_sizes, int n_in,
                              void* d_out, int out_size, void* d_ws, size_t ws_size,
                              hipStream_t stream) {
    const float* X  = (const float*)d_in[0];
    const int*   ei = (const int*)  d_in[1];
    const float* ew = (const float*)d_in[2];
    const float* H  = (const float*)d_in[3];
    const float* C  = (const float*)d_in[4];
    const float* Wx = (const float*)d_in[5];
    const float* bx = (const float*)d_in[6];
    const float* Wh = (const float*)d_in[7];
    const float* bh = (const float*)d_in[8];
    const float* wc = (const float*)d_in[9];
    const float* bg = (const float*)d_in[10];

    const int N = in_sizes[0] / 64;
    const int E = in_sizes[2];
    const int Mpad = ((N + 31) / 32) * 32;    // macro-tile is 32 rows
    const int* src = ei;
    const int* dst = ei + E;

    char* ws = (char*)d_ws;
    size_t off = 0;
    auto alloc = [&](size_t bytes) -> void* {
        void* p = (void*)(ws + off);
        off += (bytes + 255) & ~(size_t)255;
        return p;
    };
    float* deg = (float*)alloc((size_t)N * 4);
    float* nrm = (float*)alloc((size_t)E * 4);
    float* T1X = (float*)alloc((size_t)N * 64 * 4);
    float* PX  = (float*)alloc((size_t)N * 64 * 4);
    float* T1H = (float*)alloc((size_t)N * 64 * 4);
    float* PH  = (float*)alloc((size_t)N * 64 * 4);
    unsigned short* Abf = (unsigned short*)alloc((size_t)Mpad * 384 * 2);
    unsigned short* Bp  = (unsigned short*)alloc((size_t)12 * 16 * 32 * 16 * 2);
    float* Gates = (float*)alloc((size_t)Mpad * 256 * 4);

    // zero accumulation buffers (graph-capturable memset nodes)
    hipMemsetAsync(deg, 0, (size_t)N * 4, stream);
    hipMemsetAsync(T1X, 0, (size_t)N * 64 * 4, stream);
    hipMemsetAsync(PX,  0, (size_t)N * 64 * 4, stream);
    hipMemsetAsync(T1H, 0, (size_t)N * 64 * 4, stream);
    hipMemsetAsync(PH,  0, (size_t)N * 64 * 4, stream);

    const int TB = 256;
    deg_kernel <<<(E + TB - 1) / TB, TB, 0, stream>>>(ew, src, deg, E);
    norm_kernel<<<(E + TB - 1) / TB, TB, 0, stream>>>(src, dst, ew, deg, nrm, E);

    const int propBlocks = (E * 4 + TB - 1) / TB;
    prop_kernel<<<propBlocks, TB, 0, stream>>>(X,   T1X, src, dst, nrm, E);
    prop_kernel<<<propBlocks, TB, 0, stream>>>(T1X, PX,  src, dst, nrm, E);
    prop_kernel<<<propBlocks, TB, 0, stream>>>(H,   T1H, src, dst, nrm, E);
    prop_kernel<<<propBlocks, TB, 0, stream>>>(T1H, PH,  src, dst, nrm, E);

    packA_kernel<<<((size_t)Mpad * 384 + TB - 1) / TB, TB, 0, stream>>>(
        X, T1X, PX, H, T1H, PH, Abf, N, Mpad);
    packB_kernel<<<(12 * 16 * 32 * 16 + TB - 1) / TB, TB, 0, stream>>>(Wx, Wh, Bp);

    // one wave per 32x64 macro-tile; 8 waves per 256-thread block
    const int totalWaves = (Mpad / 32) * 4;
    const int gemmBlocks = (totalWaves + 7) / 8;
    gemm_wmma_kernel<<<gemmBlocks, 256, 0, stream>>>(Abf, Bp, Gates, totalWaves);

    float* Hout = (float*)d_out;
    float* Cout = Hout + (size_t)N * 64;
    gates_kernel<<<((size_t)N * 64 + TB - 1) / TB, TB, 0, stream>>>(
        Gates, C, bx, bh, bg, wc, Hout, Cout, N * 64);
}
